// CNF_36344013258794
// MI455X (gfx1250) — compile-verified, matrix operand order checked by reading, not run on hardware
//
#include <hip/hip_runtime.h>

typedef __attribute__((ext_vector_type(16))) _Float16 v16h;
typedef __attribute__((ext_vector_type(8)))  float    v8f;

#define B_ROWS 4096
#define DIM    64
#define COND   64
#define HID    512
#define IN_DIM 129
#define NINT   9

// Swizzled f16 weight storage in d_ws (units: halfs).
// Tile = 32(K) x 16(N) = 512 halfs, laid out lane-major per WMMA B layout.
#define W0_KT 5            // K padded 129 -> 160
#define W0_NT 32
#define W1_KT 16
#define W1_NT 32
#define W3_KT 16
#define W3_NT 4
#define W0H_OFF 0
#define W1H_OFF (W0H_OFF + W0_KT*W0_NT*512)
#define W2H_OFF (W1H_OFF + W1_KT*W1_NT*512)
#define W3H_OFF (W2H_OFF + W1_KT*W1_NT*512)

// Position of element with K-within-tile kk, row/col m inside a 512-half tile,
// per the ISA 16-bit A/B-matrix layout (lane halves hold interleaved K groups).
__device__ __forceinline__ int a_off(int kk, int m) {
  int h, e;
  if      (kk < 8)  { h = 0; e = kk;      }
  else if (kk < 16) { h = 1; e = kk - 8;  }
  else if (kk < 24) { h = 0; e = kk - 8;  }
  else              { h = 1; e = kk - 16; }
  return (m + 16*h)*16 + e;
}

__device__ __forceinline__ v8f wmma_f16(v16h a, v16h b, v8f c) {
  return __builtin_amdgcn_wmma_f32_16x16x32_f16(false, a, false, b,
                                                (short)0, c, false, false);
}

// Branch-free tanh: 1 - 2/(exp2(2*log2e*x)+1). Saturates to +-1, NaN propagates.
// v_exp_f32 + v_rcp_f32; avoids ocml tanh's EXEC-divergent branches.
__device__ __forceinline__ float fast_tanh(float x) {
  float e = __builtin_amdgcn_exp2f(x * 2.885390081777927f);
  return 1.0f - 2.0f * __builtin_amdgcn_rcpf(e + 1.0f);
}

// Convert f32 weight [Krows x Ncols] into swizzled f16 tiles (zero-padded).
__global__ void cnf_prep_kernel(const float* __restrict__ W, _Float16* __restrict__ dst,
                                int Krows, int Ncols, int nKt, int nNt) {
  int total = nKt * nNt * 512;
  for (int i = blockIdx.x * blockDim.x + threadIdx.x; i < total;
       i += gridDim.x * blockDim.x) {
    int e    = i & 15;
    int lane = (i >> 4) & 31;
    int tile = i >> 9;
    int nt = tile % nNt;
    int kt = tile / nNt;
    int h  = lane >> 4, nl = lane & 15;
    int kk = (e < 8) ? (e + 8*h) : (8 + e + 8*h); // e>=8 -> 16+(e-8)+8h
    int K = kt*32 + kk, N = nt*16 + nl;
    float v = (K < Krows && N < Ncols) ? W[(size_t)K * Ncols + N] : 0.0f;
    dst[i] = (_Float16)v;
  }
}

// Epilogue: bias + tanh, write activation & scaled tangent in A layout.
__device__ __forceinline__ void cnf_store_act(v8f accP, v8f accT, int nt,
                                              _Float16* ApN, _Float16* AtN,
                                              const float* __restrict__ bias,
                                              int half, int nl) {
  const int n = nt*16 + nl;
  const float bv = bias[n];
  #pragma unroll
  for (int r = 0; r < 8; ++r) {
    int m = r + 8*half;
    float u = fast_tanh(accP[r] + bv);
    int off = (n >> 5)*512 + a_off(n & 31, m);
    ApN[off] = (_Float16)u;
    AtN[off] = (_Float16)((1.0f - u*u) * accT[r]);
  }
}

// Fused primal+tangent hidden layer. 8 waves x (2 passes x 2 n-tiles) cover
// N=512. One A-pair load from LDS feeds 4 WMMAs (2 n-tiles x primal/tangent).
__device__ __forceinline__ void cnf_layer(const _Float16* __restrict__ Wt, int NT,
                                          int pKt, int tKt,
                                          const _Float16* Ap, const _Float16* At,
                                          _Float16* ApN, _Float16* AtN,
                                          const float* __restrict__ bias,
                                          int lane, int wave) {
  const int half = lane >> 4, nl = lane & 15;
  for (int p = 0; p < 2; ++p) {
    const int nt0 = wave*4 + p*2;
    v8f aP0 = {}, aP1 = {}, aT0 = {}, aT1 = {};
    #pragma unroll 2
    for (int kt = 0; kt < pKt; ++kt) {
      v16h a  = *(const v16h*)(Ap + kt*512 + lane*16);
      v16h b0 = *(const v16h*)(Wt + ((size_t)(kt*NT + nt0))*512 + lane*16);
      v16h b1 = *(const v16h*)(Wt + ((size_t)(kt*NT + nt0 + 1))*512 + lane*16);
      aP0 = wmma_f16(a, b0, aP0);
      aP1 = wmma_f16(a, b1, aP1);
      if (kt < tKt) {
        v16h at = *(const v16h*)(At + kt*512 + lane*16);
        aT0 = wmma_f16(at, b0, aT0);
        aT1 = wmma_f16(at, b1, aT1);
      }
    }
    cnf_store_act(aP0, aT0, nt0,     ApN, AtN, bias, half, nl);
    cnf_store_act(aP1, aT1, nt0 + 1, ApN, AtN, bias, half, nl);
  }
}

__global__ __launch_bounds__(256, 1)
void cnf_main_kernel(const float* __restrict__ x,   const float* __restrict__ ctx,
                     const float* __restrict__ b0,  const float* __restrict__ b1,
                     const float* __restrict__ b2,  const float* __restrict__ b3,
                     const float* __restrict__ osc, const float* __restrict__ eps,
                     const _Float16* __restrict__ wsh,
                     float* __restrict__ out_z, float* __restrict__ out_lp) {
  __shared__ __align__(32) _Float16 Apa[2][16*HID];   // primal activations (A layout)
  __shared__ __align__(32) _Float16 Ata[2][16*HID];   // tangent activations
  __shared__ float zb[16*DIM], cb[16*COND], eb[16*DIM];
  __shared__ float kzb[4][16*DIM];
  __shared__ float jvb[16*DIM];
  __shared__ float klb[4][16];
  __shared__ float lpb[16];

  const int tid  = threadIdx.x;
  const int lane = tid & 31, wave = tid >> 5;
  const int rowBase = blockIdx.x * 16;

  const _Float16* W0h = wsh + W0H_OFF;
  const _Float16* W1h = wsh + W1H_OFF;
  const _Float16* W2h = wsh + W2H_OFF;
  const _Float16* W3h = wsh + W3H_OFF;

  for (int i = tid; i < 16*DIM; i += 256) {
    int m = i >> 6, d = i & 63;
    zb[i] = x  [(size_t)(rowBase + m)*DIM  + d];
    cb[i] = ctx[(size_t)(rowBase + m)*COND + d];
  }
  if (tid < 16) lpb[tid] = 0.0f;
  const float oscale = osc[0];
  const float hstep  = -1.0f / 9.0f;

  for (int it = 0; it < NINT; ++it) {
    const float ti = 1.0f + hstep * (float)it;
    for (int st = 0; st < 4; ++st) {
      const float tcur = (st == 0) ? ti : ((st == 3) ? ti + hstep : ti + 0.5f*hstep);
      const float coef = (st == 3) ? hstep : 0.5f*hstep;
      const int  prev  = (st == 0) ? 0 : st - 1;
      __syncthreads();
      // Build primal input hdd = [z_stage | context | t | pad] as A panes 0..4.
      for (int i = tid; i < 16*160; i += 256) {
        int m = i / 160, k = i - m*160;
        float v;
        if (k < 64) {
          v = zb[m*64 + k];
          if (st != 0) v += coef * kzb[prev][m*64 + k];
        } else if (k < 128) v = cb[m*64 + (k - 64)];
        else if (k == 128)  v = tcur;
        else                v = 0.0f;
        Apa[0][(k >> 5)*512 + a_off(k & 31, m)] = (_Float16)v;
      }
      // Tangent input = eps (context/t tangents are zero); stash f32 copy.
      const float* ep = eps + (((size_t)it*4 + st) * B_ROWS + rowBase) * DIM;
      for (int i = tid; i < 16*DIM; i += 256) {
        int m = i >> 6, d = i & 63;
        float ev = ep[(size_t)m*DIM + d];
        eb[i] = ev;
        Ata[0][(d >> 5)*512 + a_off(d & 31, m)] = (_Float16)ev;
      }
      __syncthreads();
      cnf_layer(W0h, W0_NT, 5, 2,   Apa[0], Ata[0], Apa[1], Ata[1], b0, lane, wave);
      __syncthreads();
      cnf_layer(W1h, W1_NT, 16, 16, Apa[1], Ata[1], Apa[0], Ata[0], b1, lane, wave);
      __syncthreads();
      cnf_layer(W2h, W1_NT, 16, 16, Apa[0], Ata[0], Apa[1], Ata[1], b2, lane, wave);
      __syncthreads();
      // Final layer: waves 0-3 -> f (k_z), waves 4-7 -> jv.
      {
        const bool tan = (wave >= 4);
        const int  nt  = wave & 3;
        const _Float16* Asrc = tan ? Ata[1] : Apa[1];
        v8f acc = {};
        #pragma unroll 2
        for (int kt = 0; kt < 16; ++kt) {
          v16h a = *(const v16h*)(Asrc + kt*512 + lane*16);
          v16h b = *(const v16h*)(W3h + ((size_t)(kt*W3_NT + nt))*512 + lane*16);
          acc = wmma_f16(a, b, acc);
        }
        const int half = lane >> 4, nl = lane & 15;
        const int n = nt*16 + nl;
        float* dst = tan ? jvb : kzb[st];
        const float badd = tan ? 0.0f : b3[n];
        #pragma unroll
        for (int r = 0; r < 8; ++r) {
          int m = r + 8*half;
          dst[m*64 + n] = (acc[r] + badd) * oscale;
        }
      }
      __syncthreads();
      // Hutchinson divergence: kl = -sum_d jv*eps  (TSAMP==1).
      if (tid < 16) {
        float s = 0.0f;
        for (int n = 0; n < 64; ++n) s += jvb[tid*64 + n] * eb[tid*64 + n];
        klb[st][tid] = -s;
      }
    }
    __syncthreads();
    const float w6 = hstep / 6.0f;
    for (int i = tid; i < 16*DIM; i += 256)
      zb[i] += w6 * (kzb[0][i] + 2.0f*kzb[1][i] + 2.0f*kzb[2][i] + kzb[3][i]);
    if (tid < 16)
      lpb[tid] += w6 * (klb[0][tid] + 2.0f*klb[1][tid] + 2.0f*klb[2][tid] + klb[3][tid]);
  }
  __syncthreads();
  for (int i = tid; i < 16*DIM; i += 256) {
    int m = i >> 6, d = i & 63;
    out_z[(size_t)(rowBase + m)*DIM + d] = zb[i];
  }
  if (tid < 16) out_lp[rowBase + tid] = lpb[tid];
}

extern "C" void kernel_launch(void* const* d_in, const int* in_sizes, int n_in,
                              void* d_out, int out_size, void* d_ws, size_t ws_size,
                              hipStream_t stream) {
  (void)in_sizes; (void)n_in; (void)out_size; (void)ws_size;
  const float* x   = (const float*)d_in[0];
  const float* ctx = (const float*)d_in[1];
  const float* W0  = (const float*)d_in[2];
  const float* b0  = (const float*)d_in[3];
  const float* W1  = (const float*)d_in[4];
  const float* b1  = (const float*)d_in[5];
  const float* W2  = (const float*)d_in[6];
  const float* b2  = (const float*)d_in[7];
  const float* W3  = (const float*)d_in[8];
  const float* b3  = (const float*)d_in[9];
  const float* osc = (const float*)d_in[10];
  const float* eps = (const float*)d_in[11];
  _Float16* wsh = (_Float16*)d_ws;

  cnf_prep_kernel<<<256, 256, 0, stream>>>(W0, wsh + W0H_OFF, IN_DIM, HID, W0_KT, W0_NT);
  cnf_prep_kernel<<<256, 256, 0, stream>>>(W1, wsh + W1H_OFF, HID,    HID, W1_KT, W1_NT);
  cnf_prep_kernel<<<256, 256, 0, stream>>>(W2, wsh + W2H_OFF, HID,    HID, W1_KT, W1_NT);
  cnf_prep_kernel<<<256, 256, 0, stream>>>(W3, wsh + W3H_OFF, HID,    DIM, W3_KT, W3_NT);

  float* out_z  = (float*)d_out;
  float* out_lp = out_z + (size_t)B_ROWS * DIM;
  cnf_main_kernel<<<B_ROWS/16, 256, 0, stream>>>(x, ctx, b0, b1, b2, b3,
                                                 osc, eps, wsh, out_z, out_lp);
}
